// CrossModalAttention_57449482551603
// MI455X (gfx1250) — matrix operand order repaired and checked by reading, not compiled, not run on hardware
//
#include <hip/hip_runtime.h>
#include <hip/hip_bf16.h>

typedef __attribute__((ext_vector_type(16))) _Float16 v16h;
typedef __attribute__((ext_vector_type(8)))  _Float16 v8h;
typedef __attribute__((ext_vector_type(4)))  _Float16 v4h;
typedef __attribute__((ext_vector_type(8)))  float    v8f;

#define B_   8
#define C_   256
#define NSP  4096           // H*W
#define T_   512
#define SCALE 0.04419417382415922f   // 1/sqrt(512)

// ---- f16 fragment loaders, all contiguous 16B chunks -> b128 loads ----
static __device__ __forceinline__ v16h cat8(v8h a, v8h b) {
  return __builtin_shufflevector(a, b, 0,1,2,3,4,5,6,7,8,9,10,11,12,13,14,15);
}
// A-matrix 16x32: halves 0..7 -> K=kb+i, halves 8..15 -> K=kb+16+i  (kb = hi*8)
static __device__ __forceinline__ v16h fragA_ld(const _Float16* p, int kb) {
  v8h x = *(const v8h*)(p + kb);
  v8h y = *(const v8h*)(p + kb + 16);
  return cat8(x, y);
}
// B-matrix 32x16: halves 0..7 -> K=hi*8+i, halves 8..15 -> K=16+hi*8+i
static __device__ __forceinline__ v16h fragB_ld(const _Float16* p, int hi) {
  const _Float16* q = p + hi * 8;
  v8h x = *(const v8h*)(q);
  v8h y = *(const v8h*)(q + 16);
  return cat8(x, y);
}

// ---- Prologue: f32 -> f16 convert (4 elems/thread) ----
__global__ __launch_bounds__(256) void cvt_f32_f16_x4(const float* __restrict__ s,
                                                      _Float16* __restrict__ d, int n4) {
  int i = blockIdx.x * blockDim.x + threadIdx.x;
  if (i < n4) {
    float4 v = ((const float4*)s)[i];
    v4h h = { (_Float16)v.x, (_Float16)v.y, (_Float16)v.z, (_Float16)v.w };
    ((v4h*)d)[i] = h;
  }
}

// ---- Prologue: tiled transpose + convert: src f32 [R][Cc] -> dst f16 [Cc][R], per batch ----
__global__ __launch_bounds__(256) void transpose_cvt(const float* __restrict__ src,
                                                     _Float16* __restrict__ dst,
                                                     int R, int Cc) {
  __shared__ _Float16 tile[32][33];
  const int b  = blockIdx.x;
  const int r0 = blockIdx.y * 32;
  const int c0 = blockIdx.z * 32;
  const float* s = src + (size_t)b * R * Cc;
  _Float16*    d = dst + (size_t)b * R * Cc;
  const int tx = threadIdx.x & 31, ty = threadIdx.x >> 5;   // 32 x 8
#pragma unroll
  for (int i = 0; i < 32; i += 8)
    tile[ty + i][tx] = (_Float16)s[(size_t)(r0 + ty + i) * Cc + c0 + tx];
  __syncthreads();
#pragma unroll
  for (int i = 0; i < 32; i += 8)
    d[(size_t)(c0 + ty + i) * R + r0 + tx] = tile[tx][ty + i];
}

// ---- Kernel 1: scores -> softmax -> AT16 (f16, transposed) + F_s_updated ----
// 4 waves/block share one 16-row spatial tile.
__global__ __launch_bounds__(128) void xattn_scores_softmax_spatial(
    const float* __restrict__ Fs,        // f32 [B][C][N] (residual add)
    const _Float16* __restrict__ Fst16,  // [B][N][C]
    const _Float16* __restrict__ Ft16,   // [B][T][C]
    const _Float16* __restrict__ FtT16,  // [B][C][T]
    float* __restrict__ out_s,           // [B][C][N]
    _Float16* __restrict__ AT16) {       // [B][T][N]
  __shared__ float    S  [16 * T_];      // 32 KB scores
  __shared__ _Float16 S16[16 * T_];      // 16 KB softmaxed tile
  __shared__ float    red[128];
  __shared__ float    rowred[16];

  const int tid  = threadIdx.x;
  const int lane = tid & 31, wv = tid >> 5;
  const int lo = lane & 15, hi = lane >> 4, kb = hi * 8;
  const int b  = blockIdx.x >> 8;            // / (NSP/16)
  const int n0 = (blockIdx.x & 255) << 4;

  const _Float16* FstB = Fst16 + (size_t)b * NSP * C_;
  const _Float16* FtB  = Ft16  + (size_t)b * T_ * C_;
  const _Float16* FtTB = FtT16 + (size_t)b * C_ * T_;

  // Phase A: S = (F_s tile) @ F_t^T * scale. Wave wv covers t-tiles [wv*8, wv*8+8).
  v16h afrag[C_ / 32];
#pragma unroll
  for (int kk = 0; kk < C_ / 32; ++kk)
    afrag[kk] = fragA_ld(FstB + (n0 + lo) * C_ + kk * 32, kb);

  for (int tt = wv * 8; tt < wv * 8 + 8; ++tt) {
    v8f acc = {};
    const _Float16* bp = FtB + (tt * 16 + lo) * C_;
#pragma unroll
    for (int kk = 0; kk < C_ / 32; ++kk) {
      v16h bm = fragB_ld(bp + kk * 32, hi);
      acc = __builtin_amdgcn_wmma_f32_16x16x32_f16(false, afrag[kk], false, bm,
                                                   (short)0, acc, false, false);
    }
#pragma unroll
    for (int r = 0; r < 8; ++r)
      S[(r + hi * 8) * T_ + tt * 16 + lo] = acc[r] * SCALE;
  }
  __syncthreads();

  // Softmax over T: thread -> (row = tid%16, 64-col chunk q = tid/16).
  {
    const int row = tid & 15, q = tid >> 4;
    float* prow = S + row * T_ + q * 64;
    float m = -3.0e38f;
    for (int j = 0; j < 64; ++j) m = fmaxf(m, prow[j]);
    red[tid] = m;
    __syncthreads();
    if (tid < 16) {
      float mm = red[tid];
#pragma unroll
      for (int k = 1; k < 8; ++k) mm = fmaxf(mm, red[tid + k * 16]);
      rowred[tid] = mm;
    }
    __syncthreads();
    m = rowred[row];
    float s = 0.f;
    for (int j = 0; j < 64; ++j) { float e = __expf(prow[j] - m); prow[j] = e; s += e; }
    red[tid] = s;
    __syncthreads();
    if (tid < 16) {
      float ss = red[tid];
#pragma unroll
      for (int k = 1; k < 8; ++k) ss += red[tid + k * 16];
      rowred[tid] = 1.0f / ss;
    }
    __syncthreads();
    const float inv = rowred[row];
    _Float16* ATb = AT16 + (size_t)b * T_ * NSP;
    for (int j = 0; j < 64; ++j) {
      _Float16 p = (_Float16)(prow[j] * inv);
      S16[row * T_ + q * 64 + j] = p;
      ATb[(size_t)(q * 64 + j) * NSP + n0 + row] = p;   // 16 lanes -> 16 consecutive n
    }
  }
  __syncthreads();

  // Phase C: F_s_updated = F_s + A @ F_t. Wave wv covers c-tiles [wv*4, wv*4+4).
  for (int j = 0; j < 4; ++j) {
    const int ct = wv * 4 + j;
    const int c  = ct * 16 + lo;
    v8f acc = {};
#pragma unroll
    for (int kt = 0; kt < T_ / 32; ++kt) {
      v16h a  = fragA_ld(S16 + lo * T_ + kt * 32, kb);       // LDS, ds_load_b128
      v16h bm = fragB_ld(FtTB + c * T_ + kt * 32, hi);       // F_t^T contiguous over t
      acc = __builtin_amdgcn_wmma_f32_16x16x32_f16(false, a, false, bm,
                                                   (short)0, acc, false, false);
    }
#pragma unroll
    for (int r = 0; r < 8; ++r) {
      size_t idx = (size_t)b * C_ * NSP + (size_t)c * NSP + (size_t)(n0 + r + hi * 8);
      out_s[idx] = acc[r] + Fs[idx];
    }
  }
}

// ---- Kernel 2: F_t_updated = A^T @ F_s_t, one wave per 16x16 tile of [T,C] ----
__global__ __launch_bounds__(256) void xattn_text_update(
    const _Float16* __restrict__ Fs16,   // [B][C][N]
    const _Float16* __restrict__ AT16,   // [B][T][N]
    float* __restrict__ out_t) {         // [B][T][C]
  const int lane = threadIdx.x & 31;
  const int lo = lane & 15, hi = lane >> 4, kb = hi * 8;
  const int w = blockIdx.x * (blockDim.x >> 5) + (threadIdx.x >> 5);
  const int b   = w >> 9;
  const int rem = w & 511;
  const int t0 = (rem >> 4) << 4;
  const int c0 = (rem & 15) << 4;

  const _Float16* ap = AT16 + (size_t)b * T_ * NSP + (size_t)(t0 + lo) * NSP; // contiguous over n
  const _Float16* bp = Fs16 + (size_t)b * C_ * NSP + (size_t)(c0 + lo) * NSP; // contiguous over n

  v8f acc = {};
  for (int kn = 0; kn < NSP / 32; ++kn) {
    v16h a  = fragA_ld(ap + kn * 32, kb);
    v16h bm = fragB_ld(bp + kn * 32, hi);
    acc = __builtin_amdgcn_wmma_f32_16x16x32_f16(false, a, false, bm,
                                                 (short)0, acc, false, false);
  }
#pragma unroll
  for (int r = 0; r < 8; ++r)
    out_t[(size_t)b * T_ * C_ + (size_t)(t0 + r + hi * 8) * C_ + (c0 + lo)] = acc[r];
}

extern "C" void kernel_launch(void* const* d_in, const int* in_sizes, int n_in,
                              void* d_out, int out_size, void* d_ws, size_t ws_size,
                              hipStream_t stream) {
  (void)in_sizes; (void)n_in; (void)out_size; (void)ws_size;
  const float* Fs = (const float*)d_in[0];
  const float* Ft = (const float*)d_in[1];
  float* out   = (float*)d_out;
  float* out_s = out;                                  // [B,C,H,W]
  float* out_t = out + (size_t)B_ * C_ * NSP;          // [B,T,C]

  // Workspace layout (f16):
  _Float16* AT16  = (_Float16*)d_ws;                         // [B][T][N]   32 MB
  _Float16* Fs16  = AT16  + (size_t)B_ * T_ * NSP;           // [B][C][N]   16 MB
  _Float16* Fst16 = Fs16  + (size_t)B_ * C_ * NSP;           // [B][N][C]   16 MB
  _Float16* Ft16  = Fst16 + (size_t)B_ * NSP * C_;           // [B][T][C]    2 MB
  _Float16* FtT16 = Ft16  + (size_t)B_ * T_ * C_;            // [B][C][T]    2 MB

  // Prologue: one-shot converts + transposes.
  cvt_f32_f16_x4<<<(B_ * C_ * NSP) / 1024, 256, 0, stream>>>(Fs, Fs16, (B_ * C_ * NSP) / 4);
  cvt_f32_f16_x4<<<(B_ * T_ * C_) / 1024, 256, 0, stream>>>(Ft, Ft16, (B_ * T_ * C_) / 4);
  transpose_cvt<<<dim3(B_, C_ / 32, NSP / 32), 256, 0, stream>>>(Fs, Fst16, C_, NSP);
  transpose_cvt<<<dim3(B_, T_ / 32, C_ / 32), 256, 0, stream>>>(Ft, FtT16, T_, C_);

  // Kernel 1: 4 waves/block, one block per 16-row spatial tile.
  xattn_scores_softmax_spatial<<<B_ * (NSP / 16), 128, 0, stream>>>(
      Fs, Fst16, Ft16, FtT16, out_s, AT16);
  // Kernel 2: 8 waves/block, one wave per [T,C] 16x16 tile.
  xattn_text_update<<<(B_ * (T_ / 16) * (C_ / 16)) / 8, 256, 0, stream>>>(Fs16, AT16, out_t);
}
